// AVIN_6794638262660
// MI455X (gfx1250) — compile-verified, matrix operand order checked by reading, not run on hardware
//
#include <hip/hip_runtime.h>
#include <math.h>

// ---------------------------------------------------------------------------
// MI455X / gfx1250, wave32.  Heavy lifting = v_wmma_f32_16x16x32_f16 with
// double-buffered GLOBAL_LOAD_ASYNC_TO_LDS_B128 staging (ASYNCcnt).
// B=256, HW=196, CH=512.  M_conv = 50176 (=128*392),
// K_conv1 = 3456 (tap-major, 9 taps x 384ch), K_conv2 = 4608 (9 x 512).
// Convs run as implicit-im2col GEMMs with NHWC inputs and tap-major weights,
// so every 32-wide K slab is 32 contiguous channels -> vectorized b128 copies.
// ---------------------------------------------------------------------------

typedef __attribute__((ext_vector_type(16))) _Float16 v16h;
typedef __attribute__((ext_vector_type(8)))  float    v8f;

union FragH { v16h v; unsigned u[8]; };

// Async copy 16B/lane from global to LDS.  Per ISA 10.7: INST_OFFSET is added
// to BOTH the LDS destination (VDST VGPR) and the global address.
template <int OFF>
__device__ inline void async_b128(unsigned ldsOff, unsigned long long gaddr) {
    asm volatile("global_load_async_to_lds_b128 %0, %1, off offset:%2"
                 :: "v"(ldsOff), "v"(gaddr), "i"(OFF)
                 : "memory");
}
__device__ inline void wait_async0() {
    asm volatile("s_wait_asynccnt 0x0" ::: "memory");
}
// LDS byte address of a generic __shared__ pointer = low 32 bits (ISA: flat
// LDS aperture maps via addr[31:0]).
__device__ inline unsigned lds_addr(const void* p) {
    return (unsigned)(unsigned long long)p;
}

// Load a 16x32 f16 fragment from LDS per ISA 7.12.2:
// lane L: row = L&15, k-half = L>>4;
// VGPR v<4 : k = half*8 + v*2 ; VGPR v>=4 : k = 16 + half*8 + (v-4)*2.
__device__ inline v16h load_frag_lds(const _Float16* base, int rowBase,
                                     int stride, int lane) {
    const _Float16* rp = base + (rowBase + (lane & 15)) * stride;
    const int half = (lane >> 4) * 8;
    FragH f;
#pragma unroll
    for (int v = 0; v < 4; ++v)
        f.u[v] = *(const unsigned*)(rp + half + v * 2);
#pragma unroll
    for (int v = 0; v < 4; ++v)
        f.u[4 + v] = *(const unsigned*)(rp + 16 + half + v * 2);
    return f.v;
}

__device__ inline v8f wmma_f16(v16h a, v16h b, v8f c) {
    return __builtin_amdgcn_wmma_f32_16x16x32_f16(
        /*neg_a=*/false, a, /*neg_b=*/false, b,
        /*c_mod=*/(short)0, c, /*reuse_a=*/false, /*reuse_b=*/false);
}

// C/D layout: lanes 0-15 -> M=r,   N=lane ; lanes 16-31 -> M=8+r, N=lane-16.
__device__ inline void store_tile_f32(float* __restrict__ C, long ldc,
                                      long m, long n, int lane, v8f acc) {
    long col  = n + (lane & 15);
    long rowb = m + ((lane >> 4) << 3);
#pragma unroll
    for (int r = 0; r < 8; ++r) C[(rowb + r) * ldc + col] = acc[r];
}

__device__ inline void store_tile_f16_relu(_Float16* __restrict__ C, long ldc,
                                           long m, long n, int lane, v8f acc) {
    long col  = n + (lane & 15);
    long rowb = m + ((lane >> 4) << 3);
#pragma unroll
    for (int r = 0; r < 8; ++r) {
        float x = acc[r];
        C[(rowb + r) * ldc + col] = (_Float16)(x > 0.f ? x : 0.f);
    }
}

#define LDSTRIDE 40  // 32 + 8 halfs padding to dodge bank conflicts
#define TILEHALF (128 * LDSTRIDE)

// Shared WMMA core: block tile 128x128, 8 waves as 4(M) x 2(N),
// wave tile 32x64 -> 2 A frags x 4 B frags = 8 wmma per 32-K step.
struct WaveCtx {
    int lane, wm, wn;
};

__device__ inline void mma_step(const _Float16* As, const _Float16* Bs,
                                const WaveCtx& w, v8f acc[2][4]) {
    v16h aF[2], bF[4];
#pragma unroll
    for (int mm = 0; mm < 2; ++mm)
        aF[mm] = load_frag_lds(As, w.wm + mm * 16, LDSTRIDE, w.lane);
#pragma unroll
    for (int nn = 0; nn < 4; ++nn)
        bF[nn] = load_frag_lds(Bs, w.wn + nn * 16, LDSTRIDE, w.lane);
#pragma unroll
    for (int mm = 0; mm < 2; ++mm)
#pragma unroll
        for (int nn = 0; nn < 4; ++nn)
            acc[mm][nn] = wmma_f16(aF[mm], bF[nn], acc[mm][nn]);
}

__device__ inline void epilogue(float* __restrict__ Cf, _Float16* __restrict__ Ch,
                                long ldc, long m0, long n0, const WaveCtx& w,
                                v8f acc[2][4], int relu_f16) {
#pragma unroll
    for (int mm = 0; mm < 2; ++mm)
#pragma unroll
        for (int nn = 0; nn < 4; ++nn) {
            const long mB = m0 + w.wm + mm * 16;
            const long nB = n0 + w.wn + nn * 16;
            if (relu_f16) store_tile_f16_relu(Ch, ldc, mB, nB, w.lane, acc[mm][nn]);
            else          store_tile_f32(Cf, ldc, mB, nB, w.lane, acc[mm][nn]);
        }
}

// ---------------------------------------------------------------------------
// Generic NT GEMM:  C[M,N] = A[M,K] * B[N,K]^T  (A,B f16 row-major)
// Double-buffered async-to-LDS pipeline.
// ---------------------------------------------------------------------------
__global__ __launch_bounds__(256)
void avin_gemm_nt_wmma(const _Float16* __restrict__ A,
                       const _Float16* __restrict__ Bm,
                       float* __restrict__ Cf, _Float16* __restrict__ Ch,
                       int M, int N, int K, int relu_f16) {
    __shared__ _Float16 As[2][TILEHALF];
    __shared__ _Float16 Bs[2][TILEHALF];
    const int tid  = threadIdx.x;
    WaveCtx w;
    w.lane = tid & 31;
    {
        const int wave = tid >> 5;
        w.wm = (wave & 3) * 32;
        w.wn = (wave >> 2) * 64;
    }
    const long m0 = (long)blockIdx.y * 128;
    const long n0 = (long)blockIdx.x * 128;

    v8f acc[2][4] = {};

    const int row  = tid >> 1;        // 0..127 (shared by A and B loaders)
    const int part = (tid & 1) * 16;  // 0 | 16 halfs

    const _Float16* Ap = A  + (m0 + row) * (long)K + part;
    const _Float16* Bp = Bm + (n0 + row) * (long)K + part;
    unsigned ldsA[2], ldsB[2];
#pragma unroll
    for (int buf = 0; buf < 2; ++buf) {
        ldsA[buf] = lds_addr(&As[buf][row * LDSTRIDE + part]);
        ldsB[buf] = lds_addr(&Bs[buf][row * LDSTRIDE + part]);
    }

    auto issue = [&](int buf, int k0) {
        const unsigned long long ga = (unsigned long long)(Ap + k0);
        const unsigned long long gb = (unsigned long long)(Bp + k0);
        async_b128<0>(ldsA[buf], ga);
        async_b128<16>(ldsA[buf], ga);
        async_b128<0>(ldsB[buf], gb);
        async_b128<16>(ldsB[buf], gb);
    };

    issue(0, 0);
    const int S = K / 32;
    for (int s = 0; s < S; ++s) {
        const int buf = s & 1;
        wait_async0();       // my copies into `buf` have landed
        __syncthreads();     // everyone's copies landed; prev reads of buf^1 done
        if (s + 1 < S) issue(buf ^ 1, (s + 1) * 32);
        mma_step(As[buf], Bs[buf], w, acc);
    }
    epilogue(Cf, Ch, N, m0, n0, w, acc, relu_f16);
}

// ---------------------------------------------------------------------------
// 3x3 SAME conv, implicit-im2col GEMM.  M = 50176 pixel rows, N = 512.
// Input NHWC f16 [pixel, Cin]; weights tap-major f16 [512, 9*Cin]
// (k = r*Cin + c).  Every 32-K slab = 32 contiguous channels of one tap.
// Border rows zero-fill LDS directly (cannot async-load zeros).
// ---------------------------------------------------------------------------
__global__ __launch_bounds__(256)
void avin_conv3x3_wmma(const _Float16* __restrict__ In,
                       const _Float16* __restrict__ Wt,
                       float* __restrict__ Cf, _Float16* __restrict__ Ch,
                       int Cin, int relu_f16) {
    __shared__ _Float16 As[2][TILEHALF];
    __shared__ _Float16 Bs[2][TILEHALF];
    const int tid = threadIdx.x;
    WaveCtx w;
    w.lane = tid & 31;
    {
        const int wave = tid >> 5;
        w.wm = (wave & 3) * 32;
        w.wn = (wave >> 2) * 64;
    }
    const long m0 = (long)blockIdx.y * 128;
    const long n0 = (long)blockIdx.x * 128;
    const int  K  = 9 * Cin;
    const int  N  = 512;

    v8f acc[2][4] = {};

    const int row  = tid >> 1;
    const int part = (tid & 1) * 16;

    // pixel coordinates for this thread's A row
    const long m = m0 + row;
    const int b = (int)(m / 196);
    const int p = (int)(m % 196);
    const int y = p / 14, x = p % 14;

    const _Float16* Bp = Wt + (n0 + row) * (long)K + part;
    unsigned ldsA[2], ldsB[2];
#pragma unroll
    for (int buf = 0; buf < 2; ++buf) {
        ldsA[buf] = lds_addr(&As[buf][row * LDSTRIDE + part]);
        ldsB[buf] = lds_addr(&Bs[buf][row * LDSTRIDE + part]);
    }

    auto issue = [&](int buf, int rr, int kc) {
        const int y2 = y + rr / 3 - 1;
        const int x2 = x + (rr % 3) - 1;
        const bool inb = ((unsigned)y2 < 14u) && ((unsigned)x2 < 14u);
        if (inb) {
            const unsigned long long ga = (unsigned long long)(
                In + (long)(b * 196 + y2 * 14 + x2) * Cin + part + kc);
            async_b128<0>(ldsA[buf], ga);
            async_b128<16>(ldsA[buf], ga);
        } else {
            const uint4 z = {0u, 0u, 0u, 0u};
            *(uint4*)(&As[buf][row * LDSTRIDE + part])     = z;
            *(uint4*)(&As[buf][row * LDSTRIDE + part + 8]) = z;
        }
        const unsigned long long gb = (unsigned long long)(Bp + rr * Cin + kc);
        async_b128<0>(ldsB[buf], gb);
        async_b128<16>(ldsB[buf], gb);
    };

    const int S = K / 32;  // 9 * (Cin/32) steps: tap-major, channel slabs inner
    int r = 0, kc = 0;
    issue(0, 0, 0);
    for (int s = 0; s < S; ++s) {
        const int buf = s & 1;
        wait_async0();
        __syncthreads();
        int kcn = kc + 32, rn = r;
        if (kcn == Cin) { kcn = 0; rn = r + 1; }
        if (s + 1 < S) issue(buf ^ 1, rn, kcn);
        mma_step(As[buf], Bs[buf], w, acc);
        kc = kcn; r = rn;
    }
    epilogue(Cf, Ch, N, m0, n0, w, acc, relu_f16);
}

// ---------------------------------------------------------------------------
// Layout/precision prep kernels
// ---------------------------------------------------------------------------
__global__ __launch_bounds__(256)
void avin_cast_f32_f16(const float* __restrict__ X, _Float16* __restrict__ Y,
                       long n) {
    long i = (long)blockIdx.x * 256 + threadIdx.x;
    const long stride = (long)gridDim.x * 256;
    for (; i < n; i += stride) Y[i] = (_Float16)X[i];
}

// ev: f32 NCHW [B,384,196] -> f16 NHWC [B*196, 384]
__global__ __launch_bounds__(256)
void avin_nchw_to_nhwc_h(const float* __restrict__ X, _Float16* __restrict__ Y,
                         int Cin) {
    const long total = 256L * 196 * Cin;
    long i = (long)blockIdx.x * 256 + threadIdx.x;
    const long stride = (long)gridDim.x * 256;
    for (; i < total; i += stride) {
        const int c = (int)(i % Cin);
        const long pix = i / Cin;          // b*196 + p
        const int b = (int)(pix / 196);
        const int p = (int)(pix % 196);
        Y[i] = (_Float16)X[((long)b * Cin + c) * 196 + p];
    }
}

// conv weights: f32 OIHW [512, Cin, 9] -> f16 tap-major [512, 9*Cin]
__global__ __launch_bounds__(256)
void avin_reorder_conv_w(const float* __restrict__ X, _Float16* __restrict__ Y,
                         int Cin) {
    const long total = 512L * 9 * Cin;
    long i = (long)blockIdx.x * 256 + threadIdx.x;
    const long stride = (long)gridDim.x * 256;
    for (; i < total; i += stride) {
        const int c = (int)(i % Cin);
        const int r = (int)((i / Cin) % 9);
        const int n = (int)(i / ((long)9 * Cin));
        Y[i] = (_Float16)X[((long)n * Cin + c) * 9 + r];
    }
}

// Per-row L2 normalize rows of X[R,512]; f16 out (Yh) and optional f32 (Yf).
__global__ __launch_bounds__(256)
void avin_l2norm_rows(const float* __restrict__ X, _Float16* __restrict__ Yh,
                      float* __restrict__ Yf) {
    __shared__ float red[256];
    const long r = blockIdx.x;
    const int t = threadIdx.x;
    const float* xp = X + r * 512;
    float s = 0.f;
    for (int c = t; c < 512; c += 256) { float a = xp[c]; s += a * a; }
    red[t] = s;
    __syncthreads();
    for (int o = 128; o > 0; o >>= 1) {
        if (t < o) red[t] += red[t + o];
        __syncthreads();
    }
    const float inv = 1.f / fmaxf(sqrtf(red[0]), 1e-12f);
    for (int c = t; c < 512; c += 256) {
        float v = xp[c] * inv;
        Yh[r * 512 + c] = (_Float16)v;
        if (Yf) Yf[r * 512 + c] = v;
    }
}

// Masked mean pool: ind[b,c] = sum_p fv[b*196+p, c]*mask / sum(mask)
__global__ __launch_bounds__(256)
void avin_pool_ind(const float* __restrict__ fv, const int* __restrict__ masks,
                   float* __restrict__ ind) {
    __shared__ float sm[196];
    const int b = blockIdx.x;
    const int t = threadIdx.x;
    if (t < 196) sm[t] = (float)masks[b * 196 + t];
    __syncthreads();
    float ms = 0.f;
    for (int p = 0; p < 196; ++p) ms += sm[p];
    for (int c = t; c < 512; c += 256) {
        float a = 0.f;
        for (int p = 0; p < 196; ++p)
            a += fv[((long)(b * 196 + p)) * 512 + c] * sm[p];
        ind[b * 512 + c] = a / ms;
    }
}

// ---------------------------------------------------------------------------
// Exact top-k thresholds (19th largest, 99th largest == 98th smallest of 196)
// via 32-step bitwise binary search on order-preserving float keys, fused
// with the sigmoid-masked SP/SN means.  One wave per (i,j) row.
// SijT layout: [j][i*196 + p] (contiguous in p).
// ---------------------------------------------------------------------------
__global__ __launch_bounds__(256)
void avin_topk_spsn(const float* __restrict__ SijT, float* __restrict__ SP,
                    float* __restrict__ SN) {
    const int lane = threadIdx.x & 31;
    const int wave = threadIdx.x >> 5;
    const int row = blockIdx.x * 8 + wave;  // 0..65535
    const int i = row & 255;
    const int j = row >> 8;
    const float* s = SijT + (long)j * 50176 + (long)i * 196;

    float vals[7];
    unsigned key[7];
    bool ok[7];
#pragma unroll
    for (int q = 0; q < 7; ++q) {
        const int p = lane + q * 32;
        ok[q] = (p < 196);
        const float f = ok[q] ? s[p] : 0.f;
        vals[q] = f;
        const unsigned u = __float_as_uint(f);
        key[q] = ok[q] ? ((u & 0x80000000u) ? ~u : (u | 0x80000000u)) : 0u;
    }

    unsigned prefP = 0u, prefN = 0u;
    for (int bit = 31; bit >= 0; --bit) {
        const unsigned tP = prefP | (1u << bit);
        const unsigned tN = prefN | (1u << bit);
        int cP = 0, cN = 0;
#pragma unroll
        for (int q = 0; q < 7; ++q) {
            cP += (key[q] >= tP);
            cN += (key[q] >= tN);
        }
#pragma unroll
        for (int off = 16; off > 0; off >>= 1) {
            cP += __shfl_xor(cP, off);
            cN += __shfl_xor(cN, off);
        }
        if (cP >= 19) prefP = tP;   // 19th largest
        if (cN >= 99) prefN = tN;   // 99th largest == 98th smallest
    }
    const float thrP =
        __uint_as_float((prefP & 0x80000000u) ? (prefP ^ 0x80000000u) : ~prefP);
    const float thrN =
        __uint_as_float((prefN & 0x80000000u) ? (prefN ^ 0x80000000u) : ~prefN);

    const float invTS = 1.f / 0.03f;
    float spn = 0.f, spd = 0.f, snn = 0.f, snd = 0.f;
#pragma unroll
    for (int q = 0; q < 7; ++q) {
        if (ok[q]) {
            const float mp = 1.f / (1.f + __expf(-(vals[q] - thrP) * invTS));
            const float mn = 1.f - 1.f / (1.f + __expf(-(vals[q] - thrN) * invTS));
            spn += vals[q] * mp; spd += mp;
            snn += vals[q] * mn; snd += mn;
        }
    }
#pragma unroll
    for (int off = 16; off > 0; off >>= 1) {
        spn += __shfl_xor(spn, off);
        spd += __shfl_xor(spd, off);
        snn += __shfl_xor(snn, off);
        snd += __shfl_xor(snd, off);
    }
    if (lane == 0) {
        SP[i * 256 + j] = spn / spd;
        SN[i * 256 + j] = snn / snd;
    }
}

// loss1/loss2: diagonal log-softmax over [SP | SN]/TC rows + transposed rows.
__global__ __launch_bounds__(256)
void avin_loss12(const float* __restrict__ SP, const float* __restrict__ SN,
                 float* __restrict__ out) {
    __shared__ float red[256];
    const int i = threadIdx.x;
    const float invTC = 1.f / 0.07f;
    float m1 = -1e30f, m2 = -1e30f;
    for (int jj = 0; jj < 256; ++jj) {
        m1 = fmaxf(m1, fmaxf(SP[i * 256 + jj], SN[i * 256 + jj]));
        m2 = fmaxf(m2, fmaxf(SP[jj * 256 + i], SN[jj * 256 + i]));
    }
    float s1 = 0.f, s2 = 0.f;
    for (int jj = 0; jj < 256; ++jj) {
        s1 += __expf((SP[i * 256 + jj] - m1) * invTC) +
              __expf((SN[i * 256 + jj] - m1) * invTC);
        s2 += __expf((SP[jj * 256 + i] - m2) * invTC) +
              __expf((SN[jj * 256 + i] - m2) * invTC);
    }
    const float d = SP[i * 256 + i];
    red[i] = ((d - m1) * invTC - __logf(s1)) + ((d - m2) * invTC - __logf(s2));
    __syncthreads();
    for (int o = 128; o > 0; o >>= 1) {
        if (i < o) red[i] += red[i + o];
        __syncthreads();
    }
    if (i == 0) out[0] = -red[0] / 512.f;  // (loss1+loss2)/2, mean over 256
}

// dist_mat * weight * mask_mat contributions
__global__ __launch_bounds__(256)
void avin_dist_contrib(const float* __restrict__ ind, const float* __restrict__ fa,
                       const float* __restrict__ ivn, float* __restrict__ contrib) {
    const int i = blockIdx.x, j = threadIdx.x;
    const float* a = ind + i * 512;
    const float* bb = fa + j * 512;
    const float* u = ivn + i * 512;
    const float* w = ivn + j * 512;
    float d = 0.f, vv = 0.f;
    for (int c = 0; c < 512; ++c) {
        const float t = a[c] - bb[c] + 1e-6f;
        d += t * t;
        vv += u[c] * w[c];
    }
    const float wgt = (i == j) ? 1.f : -vv;
    const float mm = (i == j) ? 1.f : (-1.f / 255.f);
    contrib[i * 256 + j] = d * wgt * mm;
}

__global__ __launch_bounds__(256)
void avin_dist_loss(const float* __restrict__ contrib, float* __restrict__ out) {
    __shared__ float red[256];
    const int t = threadIdx.x;
    float rs = 0.f, cs = 0.f;
    for (int k = 0; k < 256; ++k) {
        rs += contrib[t * 256 + k];
        cs += contrib[k * 256 + t];
    }
    red[t] = fmaxf(rs + 0.6f, 0.f) + fmaxf(cs + 0.6f, 0.f);
    __syncthreads();
    for (int o = 128; o > 0; o >>= 1) {
        if (t < o) red[t] += red[t + o];
        __syncthreads();
    }
    if (t == 0) out[1] = red[0] / 512.f;  // (loss3+loss4)/2
}

// ---------------------------------------------------------------------------
// Host-side orchestration
// ---------------------------------------------------------------------------
extern "C" void kernel_launch(void* const* d_in, const int* in_sizes, int n_in,
                              void* d_out, int out_size, void* d_ws, size_t ws_size,
                              hipStream_t stream) {
    (void)in_sizes; (void)n_in; (void)out_size; (void)ws_size;
    const float* ev   = (const float*)d_in[0];  // [256,384,14,14]
    const float* ea   = (const float*)d_in[1];  // [256,2048]
    const int*   msk  = (const int*)  d_in[2];  // [256,14,14]
    const float* Wfc1 = (const float*)d_in[3];  // [512,2048]
    const float* Wfc2 = (const float*)d_in[4];  // [512,512]
    const float* Wc1  = (const float*)d_in[5];  // [512,384,3,3]
    const float* Wc2  = (const float*)d_in[6];  // [512,512,3,3]
    float* out = (float*)d_out;                 // 2 scalars

    char* ws = (char*)d_ws;
    size_t off = 0;
    auto alloc = [&](size_t bytes) -> void* {
        void* p = ws + off;
        off = (off + bytes + 255) & ~(size_t)255;
        return p;
    };
    const long MCONV = 256L * 196;  // 50176

    _Float16* ev_h   = (_Float16*)alloc(MCONV * 384 * 2);       // NHWC
    _Float16* ea_h   = (_Float16*)alloc(256L * 2048 * 2);
    _Float16* wfc1_h = (_Float16*)alloc(512L * 2048 * 2);
    _Float16* wfc2_h = (_Float16*)alloc(512L * 512 * 2);
    _Float16* wc1_h  = (_Float16*)alloc(512L * 3456 * 2);       // tap-major
    _Float16* wc2_h  = (_Float16*)alloc(512L * 4608 * 2);       // tap-major
    _Float16* fa1_h  = (_Float16*)alloc(256L * 512 * 2);
    float*    fa     = (float*)   alloc(256L * 512 * 4);
    _Float16* c1_h   = (_Float16*)alloc(MCONV * 512 * 2);       // NHWC
    float*    fv     = (float*)   alloc(MCONV * 512 * 4);
    _Float16* fvn_h  = (_Float16*)alloc(MCONV * 512 * 2);
    float*    ind    = (float*)   alloc(256L * 512 * 4);
    float*    ivn    = (float*)   alloc(256L * 512 * 4);
    _Float16* ivn_h  = (_Float16*)alloc(256L * 512 * 2);
    float*    SijT   = (float*)   alloc(256L * 50176 * 4);
    float*    SP     = (float*)   alloc(65536L * 4);
    float*    SN     = (float*)   alloc(65536L * 4);
    float*    contrib= (float*)   alloc(65536L * 4);

    // 1) precision/layout prep
    avin_nchw_to_nhwc_h<<<4096, 256, 0, stream>>>(ev, ev_h, 384);
    avin_cast_f32_f16<<<2048, 256, 0, stream>>>(ea,   ea_h,   256L * 2048);
    avin_cast_f32_f16<<<2048, 256, 0, stream>>>(Wfc1, wfc1_h, 512L * 2048);
    avin_cast_f32_f16<<<1024, 256, 0, stream>>>(Wfc2, wfc2_h, 512L * 512);
    avin_reorder_conv_w<<<2048, 256, 0, stream>>>(Wc1, wc1_h, 384);
    avin_reorder_conv_w<<<2048, 256, 0, stream>>>(Wc2, wc2_h, 512);

    // 2) Audio proj: fc1 (relu, f16 out) then fc2 (f32 out)
    avin_gemm_nt_wmma<<<dim3(512 / 128, 256 / 128), 256, 0, stream>>>(
        ea_h, wfc1_h, nullptr, fa1_h, 256, 512, 2048, 1);
    avin_gemm_nt_wmma<<<dim3(512 / 128, 256 / 128), 256, 0, stream>>>(
        fa1_h, wfc2_h, fa, nullptr, 256, 512, 512, 0);

    // 3) Visual proj: conv1 (relu, f16 NHWC out), conv2 (f32 out)
    avin_conv3x3_wmma<<<dim3(512 / 128, MCONV / 128), 256, 0, stream>>>(
        ev_h, wc1_h, nullptr, c1_h, 384, /*relu_f16=*/1);
    avin_conv3x3_wmma<<<dim3(512 / 128, MCONV / 128), 256, 0, stream>>>(
        c1_h, wc2_h, fv, nullptr, 512, /*relu_f16=*/0);

    // 4) per-pixel L2 norm of fv -> fvn_h ; masked pooling -> ind ; norm -> ivn
    avin_l2norm_rows<<<(unsigned)MCONV, 256, 0, stream>>>(fv, fvn_h, nullptr);
    avin_pool_ind<<<256, 256, 0, stream>>>(fv, msk, ind);
    avin_l2norm_rows<<<256, 256, 0, stream>>>(ind, ivn_h, ivn);

    // 5) Sij^T[j, (i,p)] = ivn[j] . fvn[(i,p)]  (p contiguous for top-k)
    avin_gemm_nt_wmma<<<dim3(50176 / 128, 256 / 128), 256, 0, stream>>>(
        ivn_h, fvn_h, SijT, nullptr, 256, 50176, 512, 0);

    // 6) exact top-k thresholds + sigmoid-masked means -> SP, SN
    avin_topk_spsn<<<65536 / 8, 256, 0, stream>>>(SijT, SP, SN);

    // 7) contrastive losses -> out[0]
    avin_loss12<<<1, 256, 0, stream>>>(SP, SN, out);

    // 8) pairwise-distance losses -> out[1]
    avin_dist_contrib<<<256, 256, 0, stream>>>(ind, fa, ivn, contrib);
    avin_dist_loss<<<1, 256, 0, stream>>>(contrib, out);
}
